// SSD_16655883174490
// MI455X (gfx1250) — matrix operand order repaired and checked by reading, not compiled, not run on hardware
//
#include <hip/hip_runtime.h>
#include <hip/hip_bf16.h>
#include <math.h>

typedef __bf16 bf16_t;
typedef __attribute__((ext_vector_type(16))) __bf16 v16bf;
typedef __attribute__((ext_vector_type(8)))  __bf16 bf16x8;
typedef __attribute__((ext_vector_type(8)))  float  v8f;

static __device__ inline v16bf zero_v16bf() {
    v16bf z;
#pragma unroll
    for (int e = 0; e < 16; ++e) z[e] = (bf16_t)0.0f;
    return z;
}

// ---------------------------------------------------------------------------
// Input: NCHW f32 -> NHWC bf16 with channel padding (C -> Cpad, zeros above C)
// ---------------------------------------------------------------------------
__global__ void ssd_nchw_to_nhwc_pad_bf16(const float* __restrict__ in, bf16_t* __restrict__ out,
                                          int Nb, int C, int Cpad, int H, int W) {
    long long i = (long long)blockIdx.x * blockDim.x + threadIdx.x;
    long long total = (long long)Nb * H * W * Cpad;
    if (i >= total) return;
    int c = (int)(i % Cpad); long long t = i / Cpad;
    int x = (int)(t % W); t /= W;
    int y = (int)(t % H);
    int n = (int)(t / H);
    float v = (c < C) ? in[(((long long)n * C + c) * H + y) * W + x] : 0.0f;
    out[i] = (bf16_t)v;
}

// ---------------------------------------------------------------------------
// Weights: OIHW f32 -> OHWI bf16, padded to (Opad rows, Ipad channels), zeros
// ---------------------------------------------------------------------------
__global__ void ssd_oihw_to_ohwi_pad_bf16(const float* __restrict__ in, bf16_t* __restrict__ out,
                                          int O, int Opad, int I, int Ipad, int KH, int KW) {
    long long i = (long long)blockIdx.x * blockDim.x + threadIdx.x;
    long long total = (long long)Opad * KH * KW * Ipad;
    if (i >= total) return;
    int ci = (int)(i % Ipad); long long t = i / Ipad;
    int kx = (int)(t % KW); t /= KW;
    int ky = (int)(t % KH);
    int o  = (int)(t / KH);
    float v = (o < O && ci < I) ? in[(((long long)o * I + ci) * KH + ky) * KW + kx] : 0.0f;
    out[i] = (bf16_t)v;
}

// ---------------------------------------------------------------------------
// Implicit-GEMM conv, NHWC bf16 activations, OHWI bf16 weights, fp32 acc.
//   Invariants (host-side padding): Cin % 32 == 0; weight tensor padded to
//   ceil(Cout/64)*64 zero rows -> the K-loop body is branch-free.
//   One wave32 computes a 32(M) x 64(N) tile: 8 accumulators, 2 A fragments.
//   The ci-block loop is software-pipelined with a rotating double buffer:
//   iteration cb issues the 12 b128 loads for cb+32 BEFORE its 8 v_wmma, so
//   a full K-step of loads is in flight during each WMMA burst.
//   All hot-loop addresses are uniform base + u32 element offsets (< 2GB);
//   __launch_bounds__(128,1) gives the allocator a spill-free budget.
//   The 4 waves of a block share tn (same weight stream) for L0/L2 reuse.
// ---------------------------------------------------------------------------
__global__ __launch_bounds__(128, 1) void ssd_conv_bf16_wmma(
    const bf16_t* __restrict__ in, const bf16_t* __restrict__ w,
    const float* __restrict__ bias, bf16_t* __restrict__ out,
    int Nb, int Cin, int H, int W, int Cout, int Ho, int Wo,
    int kh, int kw, int stride, int pad, int dil, int dorelu)
{
    const int M       = Nb * Ho * Wo;
    const int tiles_m = (M + 31) >> 5;
    const int tiles_n = (Cout + 63) >> 6;
    const int tnb = blockIdx.x % tiles_n;           // block-uniform N tile
    const int tmb = blockIdx.x / tiles_n;
    const int tm  = tmb * 4 + (threadIdx.x >> 5);   // wave-private M tile
    if (tm >= tiles_m) return;                      // wave-uniform exit

    const int lane = threadIdx.x & 31;
    const int half = lane >> 4;
    const int r    = lane & 15;

    // ---- two A-row coordinate sets (div chains once, outside hot loop)
    int nimg[2], iy0[2], ix0[2]; bool mval[2];
#pragma unroll
    for (int s = 0; s < 2; ++s) {
        int m = tm * 32 + s * 16 + r;
        mval[s] = (m < M);
        int ni = 0, oy = 0, ox = 0;
        if (mval[s]) { ni = m / (Ho * Wo); int t = m % (Ho * Wo); oy = t / Wo; ox = t % Wo; }
        nimg[s] = ni;
        iy0[s]  = oy * stride - pad;
        ix0[s]  = ox * stride - pad;
    }

    const int khw = kh * kw;
    unsigned wbase[4];                              // u32 element offsets into w
#pragma unroll
    for (int j = 0; j < 4; ++j)
        wbase[j] = (unsigned)((tnb * 64 + j * 16 + r) * khw * Cin + half * 8);

    v8f acc[2][4] = {};

    for (int ky = 0; ky < kh; ++ky) {
        int iy[2]; bool rok[2];
#pragma unroll
        for (int s = 0; s < 2; ++s) {
            iy[s]  = iy0[s] + ky * dil;
            rok[s] = mval[s] && ((unsigned)iy[s] < (unsigned)H);
        }
        for (int kx = 0; kx < kw; ++kx) {
            bool ok[2]; unsigned aoff[2];
#pragma unroll
            for (int s = 0; s < 2; ++s) {
                int ix = ix0[s] + kx * dil;
                ok[s] = rok[s] && ((unsigned)ix < (unsigned)W);
                aoff[s] = ok[s] ? (unsigned)(((nimg[s] * H + iy[s]) * W + ix) * Cin + half * 8)
                                : 0u;
            }
            const unsigned wk = (unsigned)((ky * kw + kx) * Cin);

            // fragment loader for one ci-block
            auto loadf = [&](unsigned cb, v16bf a[2], v16bf b[4]) {
#pragma unroll
                for (int s = 0; s < 2; ++s) {
                    if (ok[s]) {
                        const bf16x8 lo = *(const bf16x8*)(in + aoff[s] + cb);
                        const bf16x8 hi = *(const bf16x8*)(in + aoff[s] + cb + 16u);
                        a[s] = __builtin_shufflevector(lo, hi, 0, 1, 2, 3, 4, 5, 6, 7,
                                                       8, 9, 10, 11, 12, 13, 14, 15);
                    } else {
                        a[s] = zero_v16bf();
                    }
                }
#pragma unroll
                for (int j = 0; j < 4; ++j) {
                    const bf16_t* bb = w + (wbase[j] + wk + cb);
                    const bf16x8 blo = *(const bf16x8*)(bb);
                    const bf16x8 bhi = *(const bf16x8*)(bb + 16);
                    b[j] = __builtin_shufflevector(blo, bhi, 0, 1, 2, 3, 4, 5, 6, 7,
                                                   8, 9, 10, 11, 12, 13, 14, 15);
                }
            };
            auto compute = [&](v16bf a[2], v16bf b[4]) {
#pragma unroll
                for (int j = 0; j < 4; ++j) {
                    acc[0][j] = __builtin_amdgcn_wmma_f32_16x16x32_bf16(
                                    false, a[0], false, b[j], (short)0, acc[0][j], false, false);
                    acc[1][j] = __builtin_amdgcn_wmma_f32_16x16x32_bf16(
                                    false, a[1], false, b[j], (short)0, acc[1][j], false, false);
                }
            };

            // ---- software pipeline over ci-blocks (rotating double buffer)
            v16bf curA[2], curB[4];
            loadf(0u, curA, curB);
            for (int cb = 32; cb < Cin; cb += 32) {
                v16bf nxtA[2], nxtB[4];
                loadf((unsigned)cb, nxtA, nxtB);     // issue next K-step's loads
                compute(curA, curB);                 // overlap WMMAs with them
#pragma unroll
                for (int s = 0; s < 2; ++s) curA[s] = nxtA[s];
#pragma unroll
                for (int j = 0; j < 4; ++j) curB[j] = nxtB[j];
            }
            compute(curA, curB);                     // drain
        }
    }

    // ---- epilogue: decompose the 16 D rows once, then store 4 columns each
    unsigned doff[16]; bool dok[16];
#pragma unroll
    for (int q = 0; q < 16; ++q) {
        int s = q >> 3, i = q & 7;
        int mm = tm * 32 + s * 16 + i + 8 * half;
        dok[q] = (mm < M);
        int nim = 0, t = 0;
        if (dok[q]) { nim = mm / (Ho * Wo); t = mm % (Ho * Wo); }
        doff[q] = (unsigned)(((nim * Ho + t / Wo) * Wo + t % Wo) * Cout);
    }
#pragma unroll
    for (int j = 0; j < 4; ++j) {
        const int  ncol = tnb * 64 + j * 16 + r;
        const bool nok  = (ncol < Cout);
        float bj = nok ? bias[ncol] : 0.0f;
#pragma unroll
        for (int q = 0; q < 16; ++q) {
            int s = q >> 3, i = q & 7;
            if (dok[q] && nok) {
                float v = acc[s][j][i] + bj;
                if (dorelu) v = fmaxf(v, 0.0f);
                out[doff[q] + (unsigned)ncol] = (bf16_t)v;
            }
        }
    }
}

// ---------------------------------------------------------------------------
// Max-pool, NHWC (channel-fastest => coalesced). OOB skipped == -inf padding.
// ---------------------------------------------------------------------------
__global__ void ssd_maxpool_bf16(const bf16_t* __restrict__ in, bf16_t* __restrict__ out,
                                 int Nb, int C, int H, int W, int Ho, int Wo,
                                 int k, int s, int pad)
{
    long long i = (long long)blockIdx.x * blockDim.x + threadIdx.x;
    long long total = (long long)Nb * Ho * Wo * C;
    if (i >= total) return;
    int c = (int)(i % C); long long t = i / C;
    int xx = (int)(t % Wo); t /= Wo;
    int yy = (int)(t % Ho);
    int n  = (int)(t / Ho);
    float m = -INFINITY;
    for (int ky = 0; ky < k; ++ky) {
        int iy = yy * s - pad + ky;
        if ((unsigned)iy >= (unsigned)H) continue;
        for (int kx = 0; kx < k; ++kx) {
            int ix = xx * s - pad + kx;
            if ((unsigned)ix >= (unsigned)W) continue;
            float v = (float)in[(((long long)n * H + iy) * W + ix) * C + c];
            m = fmaxf(m, v);
        }
    }
    out[i] = (bf16_t)m;
}

// ---------------------------------------------------------------------------
// Channel L2-normalize + scale, NHWC (channels contiguous, bf16x8 vectorized)
// ---------------------------------------------------------------------------
__global__ void ssd_l2norm_scale(const bf16_t* __restrict__ in, const float* __restrict__ scale,
                                 bf16_t* __restrict__ out, long long npix, int C)
{
    long long i = (long long)blockIdx.x * blockDim.x + threadIdx.x;
    if (i >= npix) return;
    const bf16_t* base = in + i * C;
    bf16_t* obase = out + i * C;
    float s = 0.0f;
    for (int cb = 0; cb < C; cb += 8) {
        bf16x8 v = *(const bf16x8*)(base + cb);
#pragma unroll
        for (int e = 0; e < 8; ++e) { float f = (float)v[e]; s += f * f; }
    }
    float inv = 1.0f / fmaxf(sqrtf(s), 1e-12f);
    for (int cb = 0; cb < C; cb += 8) {
        bf16x8 v = *(const bf16x8*)(base + cb);
        bf16x8 o;
#pragma unroll
        for (int e = 0; e < 8; ++e) o[e] = (bf16_t)(scale[cb + e] * (float)v[e] * inv);
        *(bf16x8*)(obase + cb) = o;
    }
}

// ---------------------------------------------------------------------------
// Head scatter, NHWC inputs: cls [N,H,W,A*21], box [N,H,W,A*4] -> fp32 rows of
// d_out [N, total_rows, 25]; row = row_off + (h*W + w)*A + a. Contiguous reads.
// ---------------------------------------------------------------------------
__global__ void ssd_head_scatter(const bf16_t* __restrict__ cls, const bf16_t* __restrict__ box,
                                 float* __restrict__ out,
                                 int Nb, int A, int H, int W, int row_off, int total_rows)
{
    long long i = (long long)blockIdx.x * blockDim.x + threadIdx.x;
    long long cnt = (long long)Nb * H * W * A;
    if (i >= cnt) return;
    int a = (int)(i % A); long long t = i / A;
    int x = (int)(t % W); t /= W;
    int y = (int)(t % H);
    int n = (int)(t / H);
    long long pix = ((long long)n * H + y) * W + x;
    const bf16_t* cp = cls + pix * (A * 21) + a * 21;
    const bf16_t* bp = box + pix * (A * 4)  + a * 4;
    long long row = row_off + ((long long)y * W + x) * A + a;
    float* o = out + ((long long)n * total_rows + row) * 25;
#pragma unroll
    for (int cl = 0; cl < 21; ++cl) o[cl] = (float)cp[cl];
#pragma unroll
    for (int j = 0; j < 4; ++j) o[21 + j] = (float)bp[j];
}

// ---------------------------------------------------------------------------
// Orchestration
// ---------------------------------------------------------------------------
extern "C" void kernel_launch(void* const* d_in, const int* in_sizes, int n_in,
                              void* d_out, int out_size, void* d_ws, size_t ws_size,
                              hipStream_t stream)
{
    (void)n_in; (void)out_size;
    const int NB = in_sizes[0] / (3 * 300 * 300);
    const int TOTAL_ROWS = 4309;

    // (Cout, Cin, k) for the 35 conv pairs, in d_in order
    static const int PO[35] = { 64,64,128,128,256,256,256,512,512,512,
                                512,512,512,1024,1024,
                                256,512,128,256,128,256,128,256,
                                42,63,63,63,42,42,
                                8,12,12,12,8,8 };
    static const int PI[35] = { 3,64,64,128,128,256,256,256,512,512,
                                512,512,512,512,1024,
                                1024,256,512,128,256,128,256,128,
                                512,1024,512,256,256,256,
                                512,1024,512,256,256,256 };
    static const int PK[35] = { 3,3,3,3,3,3,3,3,3,3,
                                3,3,3,3,1,
                                1,3,1,3,1,3,1,3,
                                3,3,3,3,3,3,
                                3,3,3,3,3,3 };
    const int NPAIR = 35;

    int PIp[NPAIR], POp[NPAIR];                      // padded Cin / Cout
    for (int p = 0; p < NPAIR; ++p) {
        PIp[p] = (PI[p] % 32) ? ((PI[p] + 31) & ~31) : PI[p];   // only p0: 3->32
        POp[p] = (PO[p] + 63) & ~63;
    }

    // ---- workspace layout -------------------------------------------------
    char*  base = (char*)d_ws;
    size_t off  = 0;
    auto alloc = [&](size_t bytes) -> void* {
        void* p = base + off;
        off = (off + bytes + 255) & ~(size_t)255;
        return p;
    };
    const size_t BIG = (size_t)NB * 64 * 300 * 300;
    bf16_t* xbf    = (bf16_t*)alloc((size_t)NB * 300 * 300 * 32 * 2);   // C padded 3->32
    bf16_t* bufA   = (bf16_t*)alloc(BIG * 2);
    bf16_t* bufB   = (bf16_t*)alloc(BIG * 2);
    bf16_t* conv43 = (bf16_t*)alloc((size_t)NB * 512 * 38 * 38 * 2);
    bf16_t* f1     = (bf16_t*)alloc((size_t)NB * 512 * 38 * 38 * 2);
    bf16_t* f2     = (bf16_t*)alloc((size_t)NB * 1024 * 19 * 19 * 2);
    bf16_t* f3     = (bf16_t*)alloc((size_t)NB * 512 * 9 * 9 * 2);
    bf16_t* f4     = (bf16_t*)alloc((size_t)NB * 256 * 5 * 5 * 2);
    bf16_t* f5     = (bf16_t*)alloc((size_t)NB * 256 * 3 * 3 * 2);
    bf16_t* f6     = (bf16_t*)alloc((size_t)NB * 256 * 1 * 1 * 2);

    bf16_t* wbf[NPAIR];
    for (int p = 0; p < NPAIR; ++p)
        wbf[p] = (bf16_t*)alloc((size_t)POp[p] * PK[p] * PK[p] * PIp[p] * 2);
    if (off > ws_size) return;

    // ---- input + weight repack (deterministic every call) -----------------
    {
        long long n = (long long)NB * 300 * 300 * 32;
        ssd_nchw_to_nhwc_pad_bf16<<<(int)((n + 255) / 256), 256, 0, stream>>>(
            (const float*)d_in[0], xbf, NB, 3, 32, 300, 300);
    }
    for (int p = 0; p < NPAIR; ++p) {
        long long n = (long long)POp[p] * PK[p] * PK[p] * PIp[p];
        ssd_oihw_to_ohwi_pad_bf16<<<(int)((n + 255) / 256), 256, 0, stream>>>(
            (const float*)d_in[2 + 2 * p], wbf[p], PO[p], POp[p], PI[p], PIp[p], PK[p], PK[p]);
    }

    // ---- helpers ----------------------------------------------------------
    auto conv = [&](const bf16_t* in, int H, int W, int p,
                    int st, int pad, int dil, bf16_t* outp, int Ho, int Wo, int relu) {
        int Cin = PIp[p], Cout = PO[p], k = PK[p];
        int M = NB * Ho * Wo;
        int tiles_m = (M + 31) / 32;
        int tiles_n = (Cout + 63) / 64;
        int blocks = ((tiles_m + 3) / 4) * tiles_n;        // 4 waves/block share tn
        ssd_conv_bf16_wmma<<<blocks, 128, 0, stream>>>(
            in, wbf[p], (const float*)d_in[2 + 2 * p + 1], outp,
            NB, Cin, H, W, Cout, Ho, Wo, k, k, st, pad, dil, relu);
    };
    auto pool = [&](const bf16_t* in, int C, int H, int W, int Ho, int Wo,
                    int k, int s, int pad, bf16_t* outp) {
        long long total = (long long)NB * Ho * Wo * C;
        ssd_maxpool_bf16<<<(int)((total + 255) / 256), 256, 0, stream>>>(
            in, outp, NB, C, H, W, Ho, Wo, k, s, pad);
    };

    // ---- VGG stages 1-4 ---------------------------------------------------
    conv(xbf,  300, 300,  0, 1, 1, 1, bufA, 300, 300, 1);
    conv(bufA, 300, 300,  1, 1, 1, 1, bufB, 300, 300, 1);
    pool(bufB, 64, 300, 300, 150, 150, 2, 2, 0, bufA);
    conv(bufA, 150, 150,  2, 1, 1, 1, bufB, 150, 150, 1);
    conv(bufB, 150, 150,  3, 1, 1, 1, bufA, 150, 150, 1);
    pool(bufA, 128, 150, 150, 75, 75, 2, 2, 0, bufB);
    conv(bufB, 75, 75,  4, 1, 1, 1, bufA, 75, 75, 1);
    conv(bufA, 75, 75,  5, 1, 1, 1, bufB, 75, 75, 1);
    conv(bufB, 75, 75,  6, 1, 1, 1, bufA, 75, 75, 1);
    pool(bufA, 256, 75, 75, 38, 38, 2, 2, 0, bufB);        // ceil_mode 75 -> 38
    conv(bufB, 38, 38,  7, 1, 1, 1, bufA, 38, 38, 1);
    conv(bufA, 38, 38,  8, 1, 1, 1, bufB, 38, 38, 1);
    conv(bufB, 38, 38,  9, 1, 1, 1, conv43, 38, 38, 1);

    // ---- f1: L2-norm + scale ---------------------------------------------
    {
        long long npix = (long long)NB * 38 * 38;
        ssd_l2norm_scale<<<(int)((npix + 255) / 256), 256, 0, stream>>>(
            conv43, (const float*)d_in[1], f1, npix, 512);
    }

    // ---- conv5 + dilated fc6/fc7 -> f2 -----------------------------------
    pool(conv43, 512, 38, 38, 19, 19, 2, 2, 0, bufA);
    conv(bufA, 19, 19, 10, 1, 1, 1, bufB, 19, 19, 1);
    conv(bufB, 19, 19, 11, 1, 1, 1, bufA, 19, 19, 1);
    conv(bufA, 19, 19, 12, 1, 1, 1, bufB, 19, 19, 1);
    pool(bufB, 512, 19, 19, 19, 19, 3, 1, 1, bufA);        // k3 s1 pad1
    conv(bufA, 19, 19, 13, 1, 6, 6, bufB, 19, 19, 1);      // fc6 dil6
    conv(bufB, 19, 19, 14, 1, 0, 1, f2,   19, 19, 1);      // fc7 k1

    // ---- extras -----------------------------------------------------------
    conv(f2,   19, 19, 15, 1, 0, 1, bufA, 19, 19, 1);
    conv(bufA, 19, 19, 16, 2, 0, 1, f3,    9,  9, 1);
    conv(f3,    9,  9, 17, 1, 0, 1, bufA,  9,  9, 1);
    conv(bufA,  9,  9, 18, 2, 1, 1, f4,    5,  5, 1);
    conv(f4,    5,  5, 19, 1, 0, 1, bufA,  5,  5, 1);
    conv(bufA,  5,  5, 20, 1, 0, 1, f5,    3,  3, 1);
    conv(f5,    3,  3, 21, 1, 0, 1, bufA,  3,  3, 1);
    conv(bufA,  3,  3, 22, 1, 0, 1, f6,    1,  1, 1);

    // ---- detection heads --------------------------------------------------
    const bf16_t* feats[6] = { f1, f2, f3, f4, f5, f6 };
    const int fH[6]   = { 38, 19, 9, 5, 3, 1 };
    const int fA[6]   = { 2, 3, 3, 3, 2, 2 };
    const int roff[6] = { 0, 2888, 3971, 4214, 4289, 4307 };
    for (int i = 0; i < 6; ++i) {
        int H = fH[i], A = fA[i];
        conv(feats[i], H, H, 23 + i, 1, 1, 1, bufA, H, H, 0);   // cls head
        conv(feats[i], H, H, 29 + i, 1, 1, 1, bufB, H, H, 0);   // bbox head
        long long cnt = (long long)NB * H * H * A;
        ssd_head_scatter<<<(int)((cnt + 255) / 256), 256, 0, stream>>>(
            bufA, bufB, (float*)d_out, NB, A, H, H, roff[i], TOTAL_ROWS);
    }
}
